// TFSegformerEfficientSelfAttention_55336358641683
// MI455X (gfx1250) — matrix operand order repaired and checked by reading, not compiled, not run on hardware
//
#include <hip/hip_runtime.h>
#include <hip/hip_bf16.h>

// ---------------------------------------------------------------------------
// CDNA5 (gfx1250) implementation of Segformer efficient self-attention.
// All GEMMs use v_wmma_f32_16x16x32_bf16 (wave32 WMMA), attention is
// flash-style with LDS-staged K/V tiles and online softmax.
// ---------------------------------------------------------------------------

typedef __attribute__((ext_vector_type(16))) __bf16 v16bf;
typedef __attribute__((ext_vector_type(8)))  float  v8f;

union Frag {
    v16bf v;
    uint4 q[2];
    unsigned short s[16];
};

__device__ __forceinline__ unsigned short f2bf(float f) {
    unsigned int u = __float_as_uint(f);
    unsigned int r = u + 0x7FFFu + ((u >> 16) & 1u);   // round-to-nearest-even
    return (unsigned short)(r >> 16);
}

__device__ __forceinline__ v8f v8f_zero() {
    v8f z;
#pragma unroll
    for (int i = 0; i < 8; ++i) z[i] = 0.0f;
    return z;
}

__device__ __forceinline__ v8f wmma_bf16(v16bf a, v16bf b, v8f c) {
    return __builtin_amdgcn_wmma_f32_16x16x32_bf16(
        false, a, false, b, (short)0, c, false, false);
}

// ---------------------------------------------------------------------------
// 1) f32 -> bf16 convert, vectorized (float4 in, 4x bf16 = uint2 out)
// ---------------------------------------------------------------------------
__global__ void k_cvt_x(const float* __restrict__ x,
                        unsigned short* __restrict__ xb, int n4) {
    int i = blockIdx.x * blockDim.x + threadIdx.x;
    int stride = gridDim.x * blockDim.x;
    for (; i < n4; i += stride) {
        float4 v = *(const float4*)(x + (size_t)i * 4);
        uint2 o;
        o.x = (unsigned)f2bf(v.x) | ((unsigned)f2bf(v.y) << 16);
        o.y = (unsigned)f2bf(v.z) | ((unsigned)f2bf(v.w) << 16);
        *(uint2*)(xb + (size_t)i * 4) = o;
    }
}

// w[K][128] f32 -> wt[128][K] bf16 (transpose so B-fragments read contiguously)
__global__ void k_cvt_wT(const float* __restrict__ w,
                         unsigned short* __restrict__ wt, int K) {
    int i = blockIdx.x * blockDim.x + threadIdx.x;
    if (i >= K * 128) return;
    int k = i >> 7, n = i & 127;
    wt[(size_t)n * K + k] = f2bf(w[i]);
}

// ---------------------------------------------------------------------------
// 2) Projection GEMM: out = (A[rows,128] @ W[128,128] + bias) * scale
//    A bf16 row-major, WT bf16 [n][k]; output scattered bf16 into
//    [B, H=2, S, 64] layout (row -> (b, s), col -> (h, d)), S = 1 << logS.
//    One 16x16 output tile per wave; K accumulated 32 at a time via WMMA.
// ---------------------------------------------------------------------------
__global__ __launch_bounds__(256)
void k_proj_gemm(const unsigned short* __restrict__ A,
                 const unsigned short* __restrict__ WT,
                 const float* __restrict__ bias,
                 unsigned short* __restrict__ outB,
                 int rows, int logS, float scale) {
    int lane = threadIdx.x & 31;
    int wave = threadIdx.x >> 5;
    int tile = blockIdx.x * (blockDim.x >> 5) + wave;
    int m0 = (tile >> 3) * 16;      // 8 column tiles (128/16)
    int n0 = (tile & 7) * 16;
    if (m0 >= rows) return;
    int l16 = lane & 15, hi = lane >> 4;
    int Smask = (1 << logS) - 1;

    v8f acc = v8f_zero();
#pragma unroll
    for (int k0 = 0; k0 < 128; k0 += 32) {
        Frag a, b;
        const unsigned short* ap = A + (size_t)(m0 + l16) * 128 + k0 + 8 * hi;
        a.q[0] = *(const uint4*)ap;            // K = k0+8hi .. +7
        a.q[1] = *(const uint4*)(ap + 16);     // K = k0+16+8hi .. +7
        const unsigned short* bp = WT + (size_t)(n0 + l16) * 128 + k0 + 8 * hi;
        b.q[0] = *(const uint4*)bp;            // K = k0+8hi .. +7
        b.q[1] = *(const uint4*)(bp + 16);     // K = k0+16+8hi .. +7
        acc = wmma_bf16(a.v, b.v, acc);
    }

    int col = n0 + l16;
    float bv = bias[col];
    int hh = col >> 6, dd = col & 63;
#pragma unroll
    for (int i = 0; i < 8; ++i) {
        int r = m0 + i + 8 * hi;               // C/D layout: row = vgpr + 8*hi
        int bidx = r >> logS, sidx = r & Smask;
        float v = (acc[i] + bv) * scale;
        outB[((((size_t)bidx * 2 + hh) << logS) + sidx) * 64 + dd] = f2bf(v);
    }
}

// ---------------------------------------------------------------------------
// 3) 4x4/stride-4 conv as patch GEMM: [4096,2048] @ [2048,128] + bias -> f32
//    Patch gather: k -> (dy, dx, ci); contiguous 8-element groups stay inside
//    one channel octet so A-fragments remain two 16B loads per lane.
// ---------------------------------------------------------------------------
__global__ __launch_bounds__(256)
void k_conv_gemm(const unsigned short* __restrict__ Xbf,
                 const unsigned short* __restrict__ WsrT,
                 const float* __restrict__ bias,
                 float* __restrict__ out) {
    int lane = threadIdx.x & 31;
    int wave = threadIdx.x >> 5;
    int tile = blockIdx.x * (blockDim.x >> 5) + wave;   // 2048 tiles
    int m0 = (tile >> 3) * 16;
    int n0 = (tile & 7) * 16;
    int l16 = lane & 15, hi = lane >> 4;

    int r = m0 + l16;                       // patch row 0..4095
    int b = r >> 10, p = r & 1023;
    int oh = p >> 5, ow = p & 31;
    const unsigned short* xbase = Xbf + ((size_t)b << 21);  // b*128*128*128

    v8f acc = v8f_zero();
    for (int k0 = 0; k0 < 2048; k0 += 32) {
        Frag a, w;
        int kk = k0 + 8 * hi;
        {
            int ci = kk & 127, sp = kk >> 7;
            int dy = sp >> 2, dx = sp & 3;
            const unsigned short* ap =
                xbase + (((size_t)(oh * 4 + dy) * 128 + (ow * 4 + dx)) << 7) + ci;
            a.q[0] = *(const uint4*)ap;
        }
        {
            int kk2 = kk + 16;
            int ci = kk2 & 127, sp = kk2 >> 7;
            int dy = sp >> 2, dx = sp & 3;
            const unsigned short* ap =
                xbase + (((size_t)(oh * 4 + dy) * 128 + (ow * 4 + dx)) << 7) + ci;
            a.q[1] = *(const uint4*)ap;
        }
        const unsigned short* wp = WsrT + (size_t)(n0 + l16) * 2048 + k0 + 8 * hi;
        w.q[0] = *(const uint4*)wp;
        w.q[1] = *(const uint4*)(wp + 16);
        acc = wmma_bf16(a.v, w.v, acc);
    }

    int col = n0 + l16;
    float bv = bias[col];
#pragma unroll
    for (int i = 0; i < 8; ++i) {
        int rr = m0 + i + 8 * hi;
        out[(size_t)rr * 128 + col] = acc[i] + bv;
    }
}

// ---------------------------------------------------------------------------
// 4) LayerNorm over C=128, one wave per row, output bf16
// ---------------------------------------------------------------------------
__global__ __launch_bounds__(256)
void k_ln(const float* __restrict__ x, unsigned short* __restrict__ y,
          const float* __restrict__ gamma, const float* __restrict__ beta) {
    int row = blockIdx.x * (blockDim.x >> 5) + (threadIdx.x >> 5);
    int lane = threadIdx.x & 31;
    const float* xr = x + (size_t)row * 128;
    float4 v = *(const float4*)(xr + lane * 4);
    float s = v.x + v.y + v.z + v.w;
#pragma unroll
    for (int m = 1; m < 32; m <<= 1) s += __shfl_xor(s, m, 32);
    float mu = s * (1.0f / 128.0f);
    float d0 = v.x - mu, d1 = v.y - mu, d2 = v.z - mu, d3 = v.w - mu;
    float s2 = d0 * d0 + d1 * d1 + d2 * d2 + d3 * d3;
#pragma unroll
    for (int m = 1; m < 32; m <<= 1) s2 += __shfl_xor(s2, m, 32);
    float rstd = rsqrtf(s2 * (1.0f / 128.0f) + 1e-5f);
    int c = lane * 4;
    y[(size_t)row * 128 + c + 0] = f2bf(d0 * rstd * gamma[c + 0] + beta[c + 0]);
    y[(size_t)row * 128 + c + 1] = f2bf(d1 * rstd * gamma[c + 1] + beta[c + 1]);
    y[(size_t)row * 128 + c + 2] = f2bf(d2 * rstd * gamma[c + 2] + beta[c + 2]);
    y[(size_t)row * 128 + c + 3] = f2bf(d3 * rstd * gamma[c + 3] + beta[c + 3]);
}

// ---------------------------------------------------------------------------
// 5) Flash attention: block = 4 waves x 16 queries = 64 queries, one (b,h).
//    Loop over 32-key blocks: stage K (row-major) and V^T in LDS, QK^T via
//    WMMA, online softmax with half-wave shuffle reductions, P relayout
//    through per-wave LDS scratch (s_wait_dscnt + wave_barrier), PV via WMMA.
//    Next tile is prefetched (global_prefetch) before compute of current one.
//    1/sqrt(64) is pre-folded into Q.
// ---------------------------------------------------------------------------
__global__ __launch_bounds__(128)
void k_attn(const unsigned short* __restrict__ Qb,
            const unsigned short* __restrict__ Kb,
            const unsigned short* __restrict__ Vb,
            float* __restrict__ out) {
    __shared__ __attribute__((aligned(16))) unsigned short Ks[32][72];   // pad 8
    __shared__ __attribute__((aligned(16))) unsigned short Vst[64][40];  // V^T, pad 8
    __shared__ __attribute__((aligned(16))) unsigned short Ps[4][16][40];

    int lane = threadIdx.x & 31, wave = threadIdx.x >> 5;
    int l16 = lane & 15, hi = lane >> 4;
    int qblk = blockIdx.x & 255;       // N/64
    int bh   = blockIdx.x >> 8;        // 0..7
    int q0 = qblk * 64 + wave * 16;

    const unsigned short* Qbase = Qb + (size_t)bh * 16384 * 64;
    const unsigned short* Kbase = Kb + (size_t)bh * 1024 * 64;
    const unsigned short* Vbase = Vb + (size_t)bh * 1024 * 64;

    Frag qf[2];
#pragma unroll
    for (int f = 0; f < 2; ++f) {
        const unsigned short* qp =
            Qbase + (size_t)(q0 + l16) * 64 + f * 32 + 8 * hi;
        qf[f].q[0] = *(const uint4*)qp;
        qf[f].q[1] = *(const uint4*)(qp + 16);
    }

    float m[8], l[8];
    v8f O[4];
#pragma unroll
    for (int i = 0; i < 8; ++i) { m[i] = -3.0e38f; l[i] = 0.0f; }
#pragma unroll
    for (int j = 0; j < 4; ++j) O[j] = v8f_zero();

    for (int kb = 0; kb < 32; ++kb) {
        // --- stage K tile (32 keys x 64 dims), one uint4 per thread ---
        {
            int row = threadIdx.x >> 2, ch = threadIdx.x & 3;
            uint4 kv = *(const uint4*)(Kbase + (size_t)(kb * 32 + row) * 64 + ch * 16);
            *(uint4*)&Ks[row][ch * 16] = kv;
        }
        // --- stage V^T tile (64 dims x 32 keys) ---
        {
            int d = threadIdx.x & 63, g = threadIdx.x >> 6;
            unsigned short tmp[16];
#pragma unroll
            for (int j = 0; j < 16; ++j)
                tmp[j] = Vbase[(size_t)(kb * 32 + g * 16 + j) * 64 + d];
            *(uint4*)&Vst[d][g * 16]     = *(const uint4*)&tmp[0];
            *(uint4*)&Vst[d][g * 16 + 8] = *(const uint4*)&tmp[8];
        }
        // --- prefetch next K/V tile into cache (global_prefetch_b8) ---
        if (kb + 1 < 32) {
            int row = threadIdx.x >> 2;
            __builtin_prefetch(Kbase + (size_t)((kb + 1) * 32 + row) * 64, 0, 3);
            __builtin_prefetch(Vbase + (size_t)((kb + 1) * 32 + row) * 64, 0, 3);
        }
        __syncthreads();

        // --- S = Q(16x64) @ K^T(64x32): two 16x16 WMMA columns, K split 2 ---
        v8f sc[2];
#pragma unroll
        for (int kc = 0; kc < 2; ++kc) {
            v8f acc = v8f_zero();
#pragma unroll
            for (int f = 0; f < 2; ++f) {
                Frag bfr;
                const unsigned short* kp = &Ks[kc * 16 + l16][f * 32 + 8 * hi];
                bfr.q[0] = *(const uint4*)kp;
                bfr.q[1] = *(const uint4*)(kp + 16);
                acc = wmma_bf16(qf[f].v, bfr.v, acc);
            }
            sc[kc] = acc;
        }

        // --- online softmax per row (row = i + 8*hi), 16-lane reductions ---
#pragma unroll
        for (int i = 0; i < 8; ++i) {
            float a0 = sc[0][i], a1 = sc[1][i];
            float mx = fmaxf(a0, a1);
#pragma unroll
            for (int msk = 1; msk < 16; msk <<= 1)
                mx = fmaxf(mx, __shfl_xor(mx, msk, 32));
            float mn = fmaxf(m[i], mx);
            float so = __expf(m[i] - mn);
            float p0 = __expf(a0 - mn), p1 = __expf(a1 - mn);
            float rs = p0 + p1;
#pragma unroll
            for (int msk = 1; msk < 16; msk <<= 1) rs += __shfl_xor(rs, msk, 32);
            l[i] = l[i] * so + rs;
            m[i] = mn;
#pragma unroll
            for (int j = 0; j < 4; ++j) O[j][i] *= so;
            Ps[wave][i + 8 * hi][l16]      = f2bf(p0);
            Ps[wave][i + 8 * hi][16 + l16] = f2bf(p1);
        }
        // in-wave LDS ordering: P stores must land before A-layout reads
        asm volatile("s_wait_dscnt 0" ::: "memory");
        __builtin_amdgcn_wave_barrier();

        Frag pf;
        {
            const unsigned short* pp = &Ps[wave][l16][0];
            pf.q[0] = *(const uint4*)(pp + 8 * hi);        // K = 8hi..+7
            pf.q[1] = *(const uint4*)(pp + 16 + 8 * hi);   // K = 16+8hi..+7
        }
        // --- O(16x64) += P(16x32) @ V(32x64) ---
#pragma unroll
        for (int j = 0; j < 4; ++j) {
            Frag vf;
            const unsigned short* vp = &Vst[16 * j + l16][8 * hi];
            vf.q[0] = *(const uint4*)vp;
            vf.q[1] = *(const uint4*)(vp + 16);
            O[j] = wmma_bf16(pf.v, vf.v, O[j]);
        }
        __syncthreads();
    }

    // --- epilogue: out[b, q, h*64 + d] = O / l ---
    int b = bh >> 1, h = bh & 1;
    float inv[8];
#pragma unroll
    for (int i = 0; i < 8; ++i) inv[i] = 1.0f / l[i];
#pragma unroll
    for (int j = 0; j < 4; ++j)
#pragma unroll
        for (int i = 0; i < 8; ++i) {
            int q = q0 + i + 8 * hi;
            out[((size_t)b * 16384 + q) * 128 + h * 64 + 16 * j + l16] =
                O[j][i] * inv[i];
        }
}

// ---------------------------------------------------------------------------
// launch
// ---------------------------------------------------------------------------
extern "C" void kernel_launch(void* const* d_in, const int* in_sizes, int n_in,
                              void* d_out, int out_size, void* d_ws, size_t ws_size,
                              hipStream_t stream) {
    const float* hidden  = (const float*)d_in[0];
    const float* Wq      = (const float*)d_in[1];
    const float* bq      = (const float*)d_in[2];
    const float* Wk      = (const float*)d_in[3];
    const float* bk      = (const float*)d_in[4];
    const float* Wv      = (const float*)d_in[5];
    const float* bv      = (const float*)d_in[6];
    const float* sr_k    = (const float*)d_in[7];
    const float* sr_b    = (const float*)d_in[8];
    const float* ln_g    = (const float*)d_in[9];
    const float* ln_b    = (const float*)d_in[10];
    float* out = (float*)d_out;

    char* ws = (char*)d_ws;
    unsigned short* Xbf  = (unsigned short*)(ws + 0);          // 16 MB
    unsigned short* Qbf  = (unsigned short*)(ws + 16777216);   // 16 MB
    unsigned short* WqT  = (unsigned short*)(ws + 33554432);   // 32 KB
    unsigned short* WkT  = (unsigned short*)(ws + 33587200);
    unsigned short* WvT  = (unsigned short*)(ws + 33619968);
    unsigned short* WsrT = (unsigned short*)(ws + 33652736);   // 512 KB
    float*          xcv  = (float*)         (ws + 34177024);   // 2 MB
    unsigned short* xln  = (unsigned short*)(ws + 36274176);   // 1 MB
    unsigned short* Kbf  = (unsigned short*)(ws + 37322752);   // 1 MB
    unsigned short* Vbf  = (unsigned short*)(ws + 38371328);   // 1 MB

    // converts
    k_cvt_x<<<2048, 256, 0, stream>>>(hidden, Xbf, (4 * 16384 * 128) / 4);
    k_cvt_wT<<<64, 256, 0, stream>>>(Wq, WqT, 128);
    k_cvt_wT<<<64, 256, 0, stream>>>(Wk, WkT, 128);
    k_cvt_wT<<<64, 256, 0, stream>>>(Wv, WvT, 128);
    k_cvt_wT<<<1024, 256, 0, stream>>>(sr_k, WsrT, 2048);

    // Q projection (scale 1/sqrt(64) folded in), 65536 rows -> 4096 blocks
    k_proj_gemm<<<4096, 256, 0, stream>>>(Xbf, WqT, bq, Qbf,
                                          65536, 14, 0.125f);   // S = 16384
    // conv-as-GEMM, 4096 rows -> 256 blocks
    k_conv_gemm<<<256, 256, 0, stream>>>(Xbf, WsrT, sr_b, xcv);
    // layernorm: 4096 rows, 8 waves/block -> 512 blocks
    k_ln<<<512, 256, 0, stream>>>(xcv, xln, ln_g, ln_b);
    // K / V projections (S = 1024)
    k_proj_gemm<<<256, 256, 0, stream>>>(xln, WkT, bk, Kbf, 4096, 10, 1.0f);
    k_proj_gemm<<<256, 256, 0, stream>>>(xln, WvT, bv, Vbf, 4096, 10, 1.0f);
    // attention: 8 (b,h) x 256 query blocks
    k_attn<<<2048, 128, 0, stream>>>(Qbf, Kbf, Vbf, out);
}
